// ClassicRNN_49838800502870
// MI455X (gfx1250) — matrix operand order repaired and checked by reading, not compiled, not run on hardware
//
#include <hip/hip_runtime.h>
#include <hip/hip_bf16.h>
#include <cstdint>
#include <cstddef>

// ---------------------------------------------------------------------------
// Types for CDNA5 WMMA (wave32): 16x16x32 f16 -> f32 accumulate
// ---------------------------------------------------------------------------
typedef __attribute__((ext_vector_type(16))) _Float16 v16h;
typedef __attribute__((ext_vector_type(8)))  _Float16 h8;
typedef __attribute__((ext_vector_type(8)))  float    v8f;
typedef __attribute__((ext_vector_type(4)))  int      v4i;

typedef __attribute__((address_space(1))) void as1_void;
typedef __attribute__((address_space(3))) void as3_void;
typedef __attribute__((address_space(1))) v4i  as1_v4i;
typedef __attribute__((address_space(3))) v4i  as3_v4i;

#define D_MODEL   1024
#define BATCH     32
#define SEQ_T     512
#define TB        (BATCH * SEQ_T)     // 16384 rows for the parallel GEMMs
#define RNN_BLOCKS 64

// gfx1250 async global->LDS copy path (ASYNCcnt-tracked), with fallback.
#if defined(__has_builtin)
#  if __has_builtin(__builtin_amdgcn_global_load_async_to_lds_b128) && \
      __has_builtin(__builtin_amdgcn_s_wait_asynccnt)
#    define USE_ASYNC_LDS 1
#  endif
#endif
#ifndef USE_ASYNC_LDS
#  define USE_ASYNC_LDS 0
#endif

#if USE_ASYNC_LDS
__device__ __forceinline__ void async_copy16(const _Float16* g, _Float16* l) {
  // 16 bytes per lane, global -> LDS, tracked by ASYNCcnt.
  // Builtin signature (from hipcc diagnostic): (int4 AS1*, int4 AS3*, Ii, Ii)
  __builtin_amdgcn_global_load_async_to_lds_b128(
      (as1_v4i*)(as1_void*)(void*)g,
      (as3_v4i*)(as3_void*)(void*)l, 0, 0);
}
#endif

__device__ __forceinline__ v8f wmma16x16x32(v16h a, v16h b, v8f c) {
  // D = A(16x32) * B(32x16) + C, f32 accumulate. Codegen-verified builtin.
  return __builtin_amdgcn_wmma_f32_16x16x32_f16(false, a, false, b, (short)0, c,
                                                false, false);
}

// A fragment (16x32 f16) straight from global memory, row-major lda.
// ISA layout: lanes 0-15 -> M rows (K=0..7,16..23); lanes 16-31 same M
// rows (K=8..15,24..31). Two contiguous 16B runs per lane.
__device__ __forceinline__ v16h frag_a_global(const _Float16* __restrict__ A,
                                              int lda, int m0, int k0, int lane) {
  const int row  = m0 + (lane & 15);
  const int half = lane >> 4;
  const _Float16* p = A + (size_t)row * lda + k0 + 8 * half;
  v16h a;
#pragma unroll
  for (int c = 0; c < 8; ++c) a[c] = p[c];
#pragma unroll
  for (int c = 0; c < 8; ++c) a[c + 8] = p[c + 16];
  return a;
}

// B fragment (32x16 f16) from a K-major (K x N) weight: lane%16 = N column,
// K = 8*half+c (c<8), 16+8*half+(c-8) (c>=8), matching the A fragment.
__device__ __forceinline__ v16h frag_b_global(const _Float16* __restrict__ Bt,
                                              int ldb, int k0, int n0, int lane) {
  const _Float16* p = Bt + (size_t)(k0 + 8 * (lane >> 4)) * ldb + n0 + (lane & 15);
  v16h b;
#pragma unroll
  for (int c = 0; c < 8; ++c) b[c] = p[(size_t)c * ldb];
#pragma unroll
  for (int c = 0; c < 8; ++c) b[c + 8] = p[(size_t)(16 + c) * ldb];
  return b;
}

__device__ __forceinline__ float gelu_exact(float x) {
  return 0.5f * x * (1.0f + erff(x * 0.70710678118654752f));
}
__device__ __forceinline__ float sigmoidf_(float x) {
  return 1.0f / (1.0f + __expf(-x));
}

// Monotonic grid-wide barrier (persistent kernel). Counter is zeroed by the
// init kernel at every kernel_launch call, so replays are deterministic.
__device__ __forceinline__ void grid_sync(unsigned* counter, unsigned& gen) {
  __syncthreads();
  gen += 1;
  if (threadIdx.x == 0) {
    __threadfence();
    __hip_atomic_fetch_add(counter, 1u, __ATOMIC_ACQ_REL, __HIP_MEMORY_SCOPE_AGENT);
    const unsigned target = gen * (unsigned)gridDim.x;
    while (__hip_atomic_load(counter, __ATOMIC_ACQUIRE, __HIP_MEMORY_SCOPE_AGENT) < target) {
      __builtin_amdgcn_s_sleep(1);
    }
  }
  __syncthreads();
}

// ---------------------------------------------------------------------------
// Generic WMMA GEMM: C(MxN) = act(A(MxK) @ Bt(KxN) + bias), A/Bt f16.
// Block tile 128x128, K chunks of 32. With async path: double-buffered LDS
// filled by global_load_async_to_lds_b128, pipelined with s_wait_asynccnt.
// 8 waves as 4x2; each wave owns a 32x64 sub-tile (2x4 wmma accumulators).
// ---------------------------------------------------------------------------
template <int ACT, typename OutT>
__global__ __launch_bounds__(256) void gemm_wmma_kernel(
    const _Float16* __restrict__ A, const _Float16* __restrict__ Bt,
    const float* __restrict__ bias, OutT* __restrict__ C,
    int M, int N, int K) {
  constexpr int BM = 128, BN = 128, BK = 32;
  constexpr int NBUF = USE_ASYNC_LDS ? 2 : 1;
  __shared__ _Float16 As[NBUF][BM][BK + 8];
  __shared__ _Float16 Bs[NBUF][BK][BN + 8];

  const int tid  = threadIdx.x;
  const int lane = tid & 31;
  const int wv   = tid >> 5;
  const int wr   = wv >> 1;   // 0..3  (M direction)
  const int wc   = wv & 1;    // 0..1  (N direction)
  const int bm   = blockIdx.y * BM;
  const int bn   = blockIdx.x * BN;
  const int nk   = K / BK;

  v8f acc[2][4];
#pragma unroll
  for (int i = 0; i < 2; ++i)
#pragma unroll
    for (int j = 0; j < 4; ++j) acc[i][j] = {};

#if USE_ASYNC_LDS
  // Each thread issues 4 b128 async copies per K chunk (2 for A, 2 for B).
  // OOB M rows are clamped (not skipped) so every wave's ASYNCcnt advances
  // identically -> the constant s_wait_asynccnt(4) pipeline wait is valid.
  auto issue_chunk = [&](int kc, int buf) {
    const int k0 = kc * BK;
#pragma unroll
    for (int j = 0; j < 2; ++j) {
      const int p    = tid * 2 + j;          // 0..511 (512 x 16B per tile)
      const int rowA = p >> 2, qA = p & 3;   // A: 128 rows x 4 quarters
      int growA = bm + rowA;
      if (growA > M - 1) growA = M - 1;      // clamp, keep count uniform
      async_copy16(A + (size_t)growA * K + k0 + qA * 8,
                   &As[buf][rowA][qA * 8]);
      const int rowB = p >> 4, qB = p & 15;  // B: 32 rows x 16 pieces
      async_copy16(Bt + (size_t)(k0 + rowB) * N + bn + qB * 8,
                   &Bs[buf][rowB][qB * 8]);
    }
  };
  issue_chunk(0, 0);
#endif

  for (int kc = 0; kc < nk; ++kc) {
    const int buf = (NBUF == 2) ? (kc & 1) : 0;
#if USE_ASYNC_LDS
    if (kc + 1 < nk) {
      issue_chunk(kc + 1, buf ^ 1);
      __builtin_amdgcn_s_wait_asynccnt(4);   // chunk kc complete, kc+1 in flight
    } else {
      __builtin_amdgcn_s_wait_asynccnt(0);
    }
    __syncthreads();
#else
    {  // synchronous staging fallback
      const int rowA = tid >> 1, sa = tid & 1;
      int growA = bm + rowA;
      if (growA > M - 1) growA = M - 1;
      const _Float16* pa = A + (size_t)growA * K + kc * BK + sa * 16;
      const h8 a0 = *(const h8*)pa;
      const h8 a1 = *(const h8*)(pa + 8);
      const int rowB = tid >> 3, sbq = tid & 7;
      const _Float16* pb = Bt + (size_t)(kc * BK + rowB) * N + bn + sbq * 16;
      const h8 b0 = *(const h8*)pb;
      const h8 b1 = *(const h8*)(pb + 8);
#pragma unroll
      for (int j = 0; j < 8; ++j) {
        As[0][rowA][sa * 16 + j]      = a0[j];
        As[0][rowA][sa * 16 + 8 + j]  = a1[j];
        Bs[0][rowB][sbq * 16 + j]     = b0[j];
        Bs[0][rowB][sbq * 16 + 8 + j] = b1[j];
      }
    }
    __syncthreads();
#endif

    v16h af[2];
#pragma unroll
    for (int mt = 0; mt < 2; ++mt) {
      const int row  = wr * 32 + mt * 16 + (lane & 15);
      const int half = lane >> 4;
#pragma unroll
      for (int c = 0; c < 8; ++c) af[mt][c]     = As[buf][row][8 * half + c];
#pragma unroll
      for (int c = 0; c < 8; ++c) af[mt][c + 8] = As[buf][row][8 * half + 16 + c];
    }
#pragma unroll
    for (int nt = 0; nt < 4; ++nt) {
      v16h bf;
      const int nb0 = wc * 64 + nt * 16;
#pragma unroll
      for (int c = 0; c < 8; ++c) bf[c] = Bs[buf][8 * (lane >> 4) + c][nb0 + (lane & 15)];
#pragma unroll
      for (int c = 0; c < 8; ++c) bf[c + 8] = Bs[buf][8 * (lane >> 4) + 16 + c][nb0 + (lane & 15)];
      acc[0][nt] = wmma16x16x32(af[0], bf, acc[0][nt]);
      acc[1][nt] = wmma16x16x32(af[1], bf, acc[1][nt]);
    }
    __syncthreads();
  }

  // Epilogue: C/D layout -> lane = N + 16*(M>=8), vgpr r = M&7
#pragma unroll
  for (int mt = 0; mt < 2; ++mt) {
#pragma unroll
    for (int nt = 0; nt < 4; ++nt) {
      const int col  = bn + wc * 64 + nt * 16 + (lane & 15);
      const float bv = bias ? bias[col] : 0.0f;
      const int rbase = bm + wr * 32 + mt * 16 + ((lane >> 4) << 3);
#pragma unroll
      for (int r = 0; r < 8; ++r) {
        const int row = rbase + r;
        if (row < M) {
          float v = acc[mt][nt][r] + bv;
          if (ACT == 1) v = gelu_exact(v);
          C[(size_t)row * N + col] = (OutT)v;
        }
      }
    }
  }
}

// ---------------------------------------------------------------------------
// Persistent recurrence kernel: 64 blocks x 256 threads. Block b owns the
// 16-column strip d0 = b*16 of D=1024. Per step:
//   GRU gates (h @ Whh^T, 3 gates x 2 M-tiles, K split over 8 waves) ->
//   LayerNorm -> fc1+GELU -> fc2+residual, grid barrier between stages.
// Weight B-fragments are loop-invariant -> compiler keeps them in VGPRs.
// ---------------------------------------------------------------------------
__global__ __launch_bounds__(256) void rnn_persistent_kernel(
    const float* __restrict__ GI, const _Float16* __restrict__ Whh,
    const float* __restrict__ b_hh,
    const _Float16* __restrict__ Wfc1, const float* __restrict__ b_fc1,
    const _Float16* __restrict__ Wfc2, const float* __restrict__ b_fc2,
    const float* __restrict__ ln_g, const float* __restrict__ ln_b,
    float* __restrict__ hA, _Float16* __restrict__ hA16,
    float* __restrict__ hG, _Float16* __restrict__ hln16,
    _Float16* __restrict__ tmp16,
    unsigned* __restrict__ counter, int T) {
  __shared__ float partial[8][6][8][32];  // [wave][tile][vgpr r][lane]
  __shared__ float red_s[2][8];
  __shared__ float stat_s[2];

  const int tid  = threadIdx.x;
  const int lane = tid & 31;
  const int wv   = tid >> 5;
  const int bid  = blockIdx.x;
  const int d0   = bid * 16;
  // element-wise phase mapping: thread -> (m, c) of a 16x16 tile
  const int em    = tid >> 4;                 // 0..15
  const int ec    = tid & 15;                 // 0..15
  const int elane = ec + ((em >> 3) << 4);    // lane of element in C layout
  const int er    = em & 7;                   // vgpr index of element
  unsigned gen = 0;

  for (int t = 0; t < T; ++t) {
    // ---------------- GRU: gh = h @ Whh^T (K split across 8 waves) --------
    {
      v8f acc[6];
#pragma unroll
      for (int i = 0; i < 6; ++i) acc[i] = {};
      for (int kk = 0; kk < 4; ++kk) {
        const int k0 = wv * 128 + kk * 32;
        const v16h a0 = frag_a_global(hA16, 1024, 0, k0, lane);
        const v16h a1 = frag_a_global(hA16, 1024, 16, k0, lane);
#pragma unroll
        for (int g = 0; g < 3; ++g) {
          const v16h bf = frag_b_global(Whh, 3072, k0, g * 1024 + d0, lane);
          acc[g * 2 + 0] = wmma16x16x32(a0, bf, acc[g * 2 + 0]);
          acc[g * 2 + 1] = wmma16x16x32(a1, bf, acc[g * 2 + 1]);
        }
      }
#pragma unroll
      for (int tI = 0; tI < 6; ++tI)
#pragma unroll
        for (int r = 0; r < 8; ++r) partial[wv][tI][r][lane] = acc[tI][r];
    }
    __syncthreads();
    // gates elementwise: r,z,n -> h_new into hG
    {
#pragma unroll
      for (int mt = 0; mt < 2; ++mt) {
        const int m = mt * 16 + em;
        float ghr = 0.f, ghz = 0.f, ghn = 0.f;
#pragma unroll
        for (int w2 = 0; w2 < 8; ++w2) {
          ghr += partial[w2][0 + mt][er][elane];
          ghz += partial[w2][2 + mt][er][elane];
          ghn += partial[w2][4 + mt][er][elane];
        }
        const int col = d0 + ec;
        ghr += b_hh[col];
        ghz += b_hh[1024 + col];
        ghn += b_hh[2048 + col];
        const float* gi = GI + ((size_t)t * BATCH + m) * 3072;
        const float rr = sigmoidf_(gi[col] + ghr);
        const float zz = sigmoidf_(gi[1024 + col] + ghz);
        const float nn = tanhf(gi[2048 + col] + rr * ghn);
        const float ho = hA[m * 1024 + col];
        hG[m * 1024 + col] = (1.0f - zz) * nn + zz * ho;
      }
      // Prefetch next step's precomputed gates (L2 -> near caches) so the
      // pull overlaps this step's remaining stages. global_prefetch_b8.
      if (t + 1 < T) {
#pragma unroll
        for (int mt = 0; mt < 2; ++mt) {
          const float* gin =
              GI + ((size_t)(t + 1) * BATCH + mt * 16 + em) * 3072 + d0 + ec;
          __builtin_prefetch(gin, 0, 3);
          __builtin_prefetch(gin + 1024, 0, 3);
          __builtin_prefetch(gin + 2048, 0, 3);
        }
      }
    }
    grid_sync(counter, gen);

    // ---------------- LayerNorm (blocks 0..31 own one row each) -----------
    if (bid < BATCH) {
      const int row = bid;
      float s = 0.f, ss = 0.f;
      for (int i = tid; i < 1024; i += 256) {
        const float v = hG[row * 1024 + i];
        s += v; ss += v * v;
      }
#pragma unroll
      for (int o = 16; o > 0; o >>= 1) {
        s  += __shfl_down(s, o, 32);
        ss += __shfl_down(ss, o, 32);
      }
      if (lane == 0) { red_s[0][wv] = s; red_s[1][wv] = ss; }
      __syncthreads();
      if (tid == 0) {
        float S = 0.f, SS = 0.f;
#pragma unroll
        for (int j = 0; j < 8; ++j) { S += red_s[0][j]; SS += red_s[1][j]; }
        const float mu  = S * (1.0f / 1024.0f);
        const float var = SS * (1.0f / 1024.0f) - mu * mu;
        stat_s[0] = mu;
        stat_s[1] = rsqrtf(var + 1e-5f);
      }
      __syncthreads();
      const float mu = stat_s[0], rstd = stat_s[1];
      for (int i = tid; i < 1024; i += 256) {
        const float v = (hG[row * 1024 + i] - mu) * rstd * ln_g[i] + ln_b[i];
        hln16[row * 1024 + i] = (_Float16)v;
      }
    }
    grid_sync(counter, gen);

    // ---------------- FFN fc1 + GELU --------------------------------------
    {
      v8f acc[2];
      acc[0] = {}; acc[1] = {};
      for (int kk = 0; kk < 4; ++kk) {
        const int k0 = wv * 128 + kk * 32;
        const v16h a0 = frag_a_global(hln16, 1024, 0, k0, lane);
        const v16h a1 = frag_a_global(hln16, 1024, 16, k0, lane);
        const v16h bf = frag_b_global(Wfc1, 1024, k0, d0, lane);
        acc[0] = wmma16x16x32(a0, bf, acc[0]);
        acc[1] = wmma16x16x32(a1, bf, acc[1]);
      }
#pragma unroll
      for (int tI = 0; tI < 2; ++tI)
#pragma unroll
        for (int r = 0; r < 8; ++r) partial[wv][tI][r][lane] = acc[tI][r];
    }
    __syncthreads();
    {
#pragma unroll
      for (int mt = 0; mt < 2; ++mt) {
        const int m = mt * 16 + em;
        float sfc = 0.f;
#pragma unroll
        for (int w2 = 0; w2 < 8; ++w2) sfc += partial[w2][mt][er][elane];
        const int col = d0 + ec;
        tmp16[m * 1024 + col] = (_Float16)gelu_exact(sfc + b_fc1[col]);
      }
    }
    grid_sync(counter, gen);

    // ---------------- FFN fc2 + residual ----------------------------------
    {
      v8f acc[2];
      acc[0] = {}; acc[1] = {};
      for (int kk = 0; kk < 4; ++kk) {
        const int k0 = wv * 128 + kk * 32;
        const v16h a0 = frag_a_global(tmp16, 1024, 0, k0, lane);
        const v16h a1 = frag_a_global(tmp16, 1024, 16, k0, lane);
        const v16h bf = frag_b_global(Wfc2, 1024, k0, d0, lane);
        acc[0] = wmma16x16x32(a0, bf, acc[0]);
        acc[1] = wmma16x16x32(a1, bf, acc[1]);
      }
#pragma unroll
      for (int tI = 0; tI < 2; ++tI)
#pragma unroll
        for (int r = 0; r < 8; ++r) partial[wv][tI][r][lane] = acc[tI][r];
    }
    __syncthreads();
    {
#pragma unroll
      for (int mt = 0; mt < 2; ++mt) {
        const int m = mt * 16 + em;
        float sfc = 0.f;
#pragma unroll
        for (int w2 = 0; w2 < 8; ++w2) sfc += partial[w2][mt][er][elane];
        const int col = d0 + ec;
        const float v = hG[m * 1024 + col] + sfc + b_fc2[col];
        hA[m * 1024 + col]   = v;
        hA16[m * 1024 + col] = (_Float16)v;
      }
    }
    grid_sync(counter, gen);
  }
}

// ---------------------------------------------------------------------------
// Small utility kernels
// ---------------------------------------------------------------------------

// x (B,T,DIN) f32 -> Xh (T,B,DIN) f16 (time-major so GI rows per step are
// contiguous)
__global__ __launch_bounds__(256) void convert_x_kernel(
    const float* __restrict__ x, _Float16* __restrict__ Xh) {
  const size_t i = (size_t)blockIdx.x * 256 + threadIdx.x;  // output index
  const int t   = (int)(i >> 15);          // / (B*DIN) = 32768
  const int rem = (int)(i & 32767);
  const int b   = rem >> 10;
  const int d   = rem & 1023;
  const size_t src = ((size_t)b * SEQ_T + t) * 1024 + d;
  Xh[i] = (_Float16)x[src];
}

// W (O x I) f32 row-major -> Wt (I x O) f16 row-major (K-major weights)
__global__ __launch_bounds__(256) void transpose_w_kernel(
    const float* __restrict__ W, _Float16* __restrict__ Wt, int O, int I) {
  __shared__ float tile[32][33];
  const int tx = threadIdx.x & 31;
  const int ty = threadIdx.x >> 5;  // 0..7
  const int i0 = blockIdx.x * 32;
  const int o0 = blockIdx.y * 32;
#pragma unroll
  for (int j = 0; j < 32; j += 8) {
    const int o = o0 + ty + j, i = i0 + tx;
    tile[ty + j][tx] = (o < O && i < I) ? W[(size_t)o * I + i] : 0.0f;
  }
  __syncthreads();
#pragma unroll
  for (int j = 0; j < 32; j += 8) {
    const int i = i0 + ty + j, o = o0 + tx;
    if (i < I && o < O) Wt[(size_t)i * O + o] = (_Float16)tile[tx][ty + j];
  }
}

__global__ __launch_bounds__(256) void init_state_kernel(
    const float* __restrict__ state, float* __restrict__ hA,
    _Float16* __restrict__ hA16, unsigned* __restrict__ counter) {
  const int i = blockIdx.x * 256 + threadIdx.x;
  if (i < BATCH * 1024) {
    const float v = state[i];
    hA[i]   = v;
    hA16[i] = (_Float16)v;
  }
  if (i == 0) *counter = 0u;
}

__global__ __launch_bounds__(256) void copy_state_kernel(
    const float* __restrict__ hA, float* __restrict__ out) {
  const int i = blockIdx.x * 256 + threadIdx.x;
  if (i < BATCH * 1024) out[i] = hA[i];
}

// ---------------------------------------------------------------------------
// Launcher
// ---------------------------------------------------------------------------
extern "C" void kernel_launch(void* const* d_in, const int* in_sizes, int n_in,
                              void* d_out, int out_size, void* d_ws,
                              size_t ws_size, hipStream_t stream) {
  (void)in_sizes; (void)n_in; (void)out_size; (void)ws_size;

  const float* x      = (const float*)d_in[0];   // (B,T,DIN)
  const float* state  = (const float*)d_in[1];   // (B,D)
  const float* w_in   = (const float*)d_in[2];   // (D,DIN)
  const float* b_in   = (const float*)d_in[3];
  const float* w_ih   = (const float*)d_in[4];   // (3D,D)
  const float* b_ih   = (const float*)d_in[5];
  const float* w_hh   = (const float*)d_in[6];   // (3D,D)
  const float* b_hh   = (const float*)d_in[7];
  const float* ln1_g  = (const float*)d_in[8];
  const float* ln1_b  = (const float*)d_in[9];
  const float* fc1_w  = (const float*)d_in[10];  // (D,D)
  const float* fc1_b  = (const float*)d_in[11];
  const float* fc2_w  = (const float*)d_in[12];  // (D,D)
  const float* fc2_b  = (const float*)d_in[13];
  // d_in[14], d_in[15]: ln2 (unused by reference forward)
  const float* w_out  = (const float*)d_in[16];  // (DOUT,D)
  const float* b_out  = (const float*)d_in[17];
  float* out = (float*)d_out;                    // [y (32x1024) | h (32x1024)]

  // Workspace carve
  size_t off = 0;
  char* base = (char*)d_ws;
  auto carve = [&](size_t bytes) -> void* {
    void* r = base + off;
    off += (bytes + 255) & ~(size_t)255;
    return r;
  };
  unsigned*  counter = (unsigned*)carve(256);
  _Float16*  Xh      = (_Float16*)carve((size_t)TB * 1024 * 2);
  _Float16*  WinT    = (_Float16*)carve((size_t)1024 * 1024 * 2);
  _Float16*  WihT    = (_Float16*)carve((size_t)1024 * 3072 * 2);
  _Float16*  WhhT    = (_Float16*)carve((size_t)1024 * 3072 * 2);
  _Float16*  Fc1T    = (_Float16*)carve((size_t)1024 * 1024 * 2);
  _Float16*  Fc2T    = (_Float16*)carve((size_t)1024 * 1024 * 2);
  _Float16*  WoutT   = (_Float16*)carve((size_t)1024 * 1024 * 2);
  _Float16*  Xemb    = (_Float16*)carve((size_t)TB * 1024 * 2);
  float*     GI      = (float*)carve((size_t)TB * 3072 * 4);
  float*     hA      = (float*)carve((size_t)BATCH * 1024 * 4);
  float*     hG      = (float*)carve((size_t)BATCH * 1024 * 4);
  _Float16*  hA16    = (_Float16*)carve((size_t)BATCH * 1024 * 2);
  _Float16*  hln16   = (_Float16*)carve((size_t)BATCH * 1024 * 2);
  _Float16*  tmp16   = (_Float16*)carve((size_t)BATCH * 1024 * 2);

  // 1) f32 -> f16 conversions / weight transposes (K-major for B fragments)
  convert_x_kernel<<<(TB * 1024) / 256, 256, 0, stream>>>(x, Xh);
  transpose_w_kernel<<<dim3(1024 / 32, 1024 / 32), 256, 0, stream>>>(w_in, WinT, 1024, 1024);
  transpose_w_kernel<<<dim3(1024 / 32, 3072 / 32), 256, 0, stream>>>(w_ih, WihT, 3072, 1024);
  transpose_w_kernel<<<dim3(1024 / 32, 3072 / 32), 256, 0, stream>>>(w_hh, WhhT, 3072, 1024);
  transpose_w_kernel<<<dim3(1024 / 32, 1024 / 32), 256, 0, stream>>>(fc1_w, Fc1T, 1024, 1024);
  transpose_w_kernel<<<dim3(1024 / 32, 1024 / 32), 256, 0, stream>>>(fc2_w, Fc2T, 1024, 1024);
  transpose_w_kernel<<<dim3(1024 / 32, 1024 / 32), 256, 0, stream>>>(w_out, WoutT, 1024, 1024);
  init_state_kernel<<<(BATCH * 1024) / 256, 256, 0, stream>>>(state, hA, hA16, counter);

  // 2) Parallel precompute: Xemb = Xh @ WinT + b_in ; GI = Xemb @ WihT + b_ih
  gemm_wmma_kernel<0, _Float16><<<dim3(1024 / 128, TB / 128), 256, 0, stream>>>(
      Xh, WinT, b_in, Xemb, TB, 1024, 1024);
  gemm_wmma_kernel<0, float><<<dim3(3072 / 128, TB / 128), 256, 0, stream>>>(
      Xemb, WihT, b_ih, GI, TB, 3072, 1024);

  // 3) Sequential recurrence: one persistent kernel, grid barrier per stage
  rnn_persistent_kernel<<<RNN_BLOCKS, 256, 0, stream>>>(
      GI, WhhT, b_hh, Fc1T, fc1_b, Fc2T, fc2_b, ln1_g, ln1_b,
      hA, hA16, hG, hln16, tmp16, counter, SEQ_T);

  // 4) Output projection: y = h @ WoutT + b_out, then copy new_state
  gemm_wmma_kernel<0, float><<<dim3(1024 / 128, 1), 256, 0, stream>>>(
      hA16, WoutT, b_out, out, BATCH, 1024, 1024);
  copy_state_kernel<<<(BATCH * 1024) / 256, 256, 0, stream>>>(hA, out + BATCH * 1024);
}
